// BottleneckS4D_6356551598500
// MI455X (gfx1250) — compile-verified, matrix operand order checked by reading, not run on hardware
//
#include <hip/hip_runtime.h>
#include <hip/hip_bf16.h>
#include <math.h>

// ---------- plain vector types (safe inside unions) ----------
typedef __bf16  v16bf __attribute__((ext_vector_type(16)));
typedef float   v8f   __attribute__((ext_vector_type(8)));
typedef float   f32x4 __attribute__((ext_vector_type(4)));
typedef unsigned int u32x4 __attribute__((ext_vector_type(4)));
typedef unsigned int u32x2 __attribute__((ext_vector_type(2)));

#define B_DIM   4
#define T_DIM   64
#define H_DIM   1280
#define N2_DIM  32
#define M_DIM   256          // B*T
#define K_DIM   150528       // 224*224*3
#define KC      3072         // K chunk (49 chunks)
#define NKC     49
#define BK      64           // K step per LDS stage
#define NSTEP   (KC / BK)    // 48
#define LDP     72           // padded LDS row (bf16); 144B rows keep 16B align + bank spread
#define RH      64
#define NC      60

// ---------------- kernel 0: u <- bias ----------------
__global__ __launch_bounds__(256) void init_u_kernel(const float* __restrict__ bb,
                                                     float* __restrict__ u) {
    int i = blockIdx.x * 256 + threadIdx.x;
    if (i < M_DIM * H_DIM) u[i] = bb[i % H_DIM];
}

// ---------------- kernel 1: u += X @ Wb^T  (split-K, bf16 WMMA, double-buffered) ----
// grid = (2, 10, 49), block = 256 (8 waves). Tile 128x128, K chunk 3072.
// Pipeline per iteration: [issue global loads it+1] -> [WMMA on LDS buf(it)]
//                      -> [cvt+store into buf(it+1)] -> one barrier.
__global__ __launch_bounds__(256) void gemm_u_kernel(const float* __restrict__ X,
                                                     const float* __restrict__ Wb,
                                                     float* __restrict__ u) {
    __shared__ __bf16 Abf[2][128][LDP];
    __shared__ __bf16 Bbf[2][128][LDP];

    const int tid  = threadIdx.x;
    const int lane = tid & 31;
    const int wave = tid >> 5;
    const int wm   = wave & 1;        // 2 wave rows  -> 64 M each
    const int wn   = wave >> 1;       // 4 wave cols  -> 32 N each
    const int bm   = blockIdx.x;
    const int bn   = blockIdx.y;
    const size_t kc0 = (size_t)blockIdx.z * KC;

    const float* Ag = X  + (size_t)(bm * 128) * K_DIM + kc0;
    const float* Bg = Wb + (size_t)(bn * 128) * K_DIM + kc0;

    // per-thread staging: 8 float4 segments of A tile + 8 of B tile (128x64 f32 each)
    const int srow = tid >> 4;        // shared by all 8 segments? no: derived per i below
    (void)srow;

    v8f acc[4][2];
#pragma unroll
    for (int i = 0; i < 4; ++i)
#pragma unroll
        for (int j = 0; j < 2; ++j) acc[i][j] = (v8f)0.0f;

    const int sel  = (lane >> 4) * 8;   // ISA 16-bit A layout: hi lanes hold K+8
    const int mrow = lane & 15;

    f32x4 ra[8], rb[8];

    // ---- prologue: load + convert stage 0 into buffer 0 ----
#pragma unroll
    for (int i = 0; i < 8; ++i) {
        int idx = tid + i * 256;
        int row = idx >> 4;
        int kq  = idx & 15;
        ra[i] = *(const f32x4*)(Ag + (size_t)row * K_DIM + kq * 4);
        rb[i] = *(const f32x4*)(Bg + (size_t)row * K_DIM + kq * 4);
    }
#pragma unroll
    for (int i = 0; i < 8; ++i) {
        int idx = tid + i * 256;
        int row = idx >> 4;
        int kq  = idx & 15;
        union { __bf16 h[4]; u32x2 w; } pa, pb;
        pa.h[0] = (__bf16)ra[i].x; pa.h[1] = (__bf16)ra[i].y;
        pa.h[2] = (__bf16)ra[i].z; pa.h[3] = (__bf16)ra[i].w;
        pb.h[0] = (__bf16)rb[i].x; pb.h[1] = (__bf16)rb[i].y;
        pb.h[2] = (__bf16)rb[i].z; pb.h[3] = (__bf16)rb[i].w;
        *(u32x2*)&Abf[0][row][kq * 4] = pa.w;
        *(u32x2*)&Bbf[0][row][kq * 4] = pb.w;
    }
    __syncthreads();

    for (int it = 0; it < NSTEP; ++it) {
        const int cur = it & 1;
        const int nxt = cur ^ 1;
        const int ko_next = (it + 1) * BK;
        const bool has_next = (it + 1) < NSTEP;

        // ---- (a) issue next-stage global loads early; they fly under the WMMAs ----
        if (has_next) {
#pragma unroll
            for (int i = 0; i < 8; ++i) {
                int idx = tid + i * 256;
                int row = idx >> 4;
                int kq  = idx & 15;
                ra[i] = *(const f32x4*)(Ag + (size_t)row * K_DIM + ko_next + kq * 4);
                rb[i] = *(const f32x4*)(Bg + (size_t)row * K_DIM + ko_next + kq * 4);
            }
        }

        // ---- (b) 2 WMMA K-steps of 32 from current buffer ----
#pragma unroll
        for (int kk = 0; kk < BK; kk += 32) {
            union Frag { u32x4 q[2]; v16bf v; };
            Frag a[4], b[2];
#pragma unroll
            for (int mi = 0; mi < 4; ++mi) {
                int r = wm * 64 + mi * 16 + mrow;
                a[mi].q[0] = *(const u32x4*)&Abf[cur][r][kk + sel];
                a[mi].q[1] = *(const u32x4*)&Abf[cur][r][kk + 16 + sel];
            }
#pragma unroll
            for (int ni = 0; ni < 2; ++ni) {
                int r = wn * 32 + ni * 16 + mrow;
                b[ni].q[0] = *(const u32x4*)&Bbf[cur][r][kk + sel];
                b[ni].q[1] = *(const u32x4*)&Bbf[cur][r][kk + 16 + sel];
            }
#pragma unroll
            for (int mi = 0; mi < 4; ++mi)
#pragma unroll
                for (int ni = 0; ni < 2; ++ni)
                    acc[mi][ni] = __builtin_amdgcn_wmma_f32_16x16x32_bf16(
                        false, a[mi].v, false, b[ni].v,
                        (short)0, acc[mi][ni], false, false);
        }

        // ---- (c) convert + store next stage into the other buffer ----
        if (has_next) {
#pragma unroll
            for (int i = 0; i < 8; ++i) {
                int idx = tid + i * 256;
                int row = idx >> 4;
                int kq  = idx & 15;
                union { __bf16 h[4]; u32x2 w; } pa, pb;
                pa.h[0] = (__bf16)ra[i].x; pa.h[1] = (__bf16)ra[i].y;
                pa.h[2] = (__bf16)ra[i].z; pa.h[3] = (__bf16)ra[i].w;
                pb.h[0] = (__bf16)rb[i].x; pb.h[1] = (__bf16)rb[i].y;
                pb.h[2] = (__bf16)rb[i].z; pb.h[3] = (__bf16)rb[i].w;
                *(u32x2*)&Abf[nxt][row][kq * 4] = pa.w;
                *(u32x2*)&Bbf[nxt][row][kq * 4] = pb.w;
            }
        }

        // ---- (d) single barrier: cur fully read by all, nxt fully written ----
        __syncthreads();
    }

    // ---- split-K accumulate into u (C layout: VGPR r -> M=r / r+8 by lane half) ----
#pragma unroll
    for (int mi = 0; mi < 4; ++mi)
#pragma unroll
        for (int ni = 0; ni < 2; ++ni)
#pragma unroll
            for (int r = 0; r < 8; ++r) {
                int m = bm * 128 + wm * 64 + mi * 16 + r + ((lane >> 4) * 8);
                int n = bn * 128 + wn * 32 + ni * 16 + (lane & 15);
                atomicAdd(&u[(size_t)m * H_DIM + n], acc[mi][ni][r]);
            }
}

// ---------------- kernel 2: S4D conv @ l=63 + D skip + exact GELU ----------------
__global__ __launch_bounds__(256) void s4d_conv_kernel(const float* __restrict__ u,
                                                       const float* __restrict__ log_dt,
                                                       const float* __restrict__ C,
                                                       const float* __restrict__ log_A_real,
                                                       const float* __restrict__ Dv,
                                                       float* __restrict__ y63) {
    int h = blockIdx.x * 256 + threadIdx.x;
    if (h >= H_DIM) return;

    float dt = expf(log_dt[h]);
    float Cb[N2_DIM], rr[N2_DIM], e[N2_DIM];
#pragma unroll
    for (int n = 0; n < N2_DIM; ++n) {
        float A   = -expf(log_A_real[h * N2_DIM + n]);
        float dtA = A * dt;
        float r   = expf(dtA);
        rr[n] = r;
        Cb[n] = C[h * N2_DIM + n] * (r - 1.0f) / A;
        e[n]  = 1.0f;
    }

    float acc[B_DIM] = {0.f, 0.f, 0.f, 0.f};
    for (int m = 0; m < T_DIM; ++m) {
        float km = 0.f;
#pragma unroll
        for (int n = 0; n < N2_DIM; ++n) km += Cb[n] * e[n];
        km *= 2.0f;
#pragma unroll
        for (int b = 0; b < B_DIM; ++b)
            acc[b] += km * u[(size_t)(b * T_DIM + (T_DIM - 1 - m)) * H_DIM + h];
#pragma unroll
        for (int n = 0; n < N2_DIM; ++n) e[n] *= rr[n];
    }

#pragma unroll
    for (int b = 0; b < B_DIM; ++b) {
        float x = acc[b] + Dv[h] * u[(size_t)(b * T_DIM + T_DIM - 1) * H_DIM + h];
        float g = 0.5f * x * (1.0f + erff(x * 0.70710678118654752f));  // exact GELU
        y63[b * H_DIM + h] = g;
    }
}

// ---------------- kernel 3: z = Wc @ y63 + bc ; GLU ----------------
__global__ __launch_bounds__(256) void zglu_kernel(const float* __restrict__ Wc,
                                                   const float* __restrict__ bc,
                                                   const float* __restrict__ y63,
                                                   float* __restrict__ yglu) {
    int o = blockIdx.x * 256 + threadIdx.x;
    if (o >= H_DIM) return;
    const float* wa = Wc + (size_t)o * H_DIM;
    const float* wg = Wc + (size_t)(o + H_DIM) * H_DIM;
    float aa[B_DIM] = {0.f, 0.f, 0.f, 0.f};
    float gg[B_DIM] = {0.f, 0.f, 0.f, 0.f};
    for (int h = 0; h < H_DIM; ++h) {
        float a = wa[h], g = wg[h];
#pragma unroll
        for (int b = 0; b < B_DIM; ++b) {
            float y = y63[b * H_DIM + h];
            aa[b] += a * y;
            gg[b] += g * y;
        }
    }
#pragma unroll
    for (int b = 0; b < B_DIM; ++b) {
        float a = aa[b] + bc[o];
        float g = gg[b] + bc[o + H_DIM];
        yglu[b * H_DIM + o] = a * (1.0f / (1.0f + expf(-g)));
    }
}

// ---------------- kernel 4: classifier head ----------------
__global__ __launch_bounds__(256) void head_kernel(const float* __restrict__ yglu,
                                                   const float* __restrict__ W1,
                                                   const float* __restrict__ b1,
                                                   const float* __restrict__ W2,
                                                   const float* __restrict__ b2,
                                                   float* __restrict__ out) {
    __shared__ float hb[B_DIM][RH];
    int t = threadIdx.x;
    {   // 256 threads = 4 batches x 64 hidden
        int b = t >> 6, r = t & 63;
        float acc = b1[r];
        const float* w = W1 + (size_t)r * H_DIM;
        for (int h = 0; h < H_DIM; ++h) acc += w[h] * yglu[b * H_DIM + h];
        hb[b][r] = fmaxf(acc, 0.0f);
    }
    __syncthreads();
    if (t < B_DIM * NC) {
        int b = t / NC, c = t % NC;
        float acc = b2[c];
#pragma unroll
        for (int r = 0; r < RH; ++r) acc += W2[c * RH + r] * hb[b][r];
        out[b * NC + c] = acc;
    }
}

extern "C" void kernel_launch(void* const* d_in, const int* in_sizes, int n_in,
                              void* d_out, int out_size, void* d_ws, size_t ws_size,
                              hipStream_t stream) {
    const float* x          = (const float*)d_in[0];
    const float* Wb         = (const float*)d_in[1];
    const float* bb         = (const float*)d_in[2];
    const float* log_dt     = (const float*)d_in[3];
    const float* C          = (const float*)d_in[4];
    const float* log_A_real = (const float*)d_in[5];
    const float* Dv         = (const float*)d_in[6];
    const float* Wc         = (const float*)d_in[7];
    const float* bc         = (const float*)d_in[8];
    const float* W1         = (const float*)d_in[9];
    const float* b1         = (const float*)d_in[10];
    const float* W2         = (const float*)d_in[11];
    const float* b2         = (const float*)d_in[12];
    float* out = (float*)d_out;

    float* u    = (float*)d_ws;                 // 256*1280 f32
    float* y63  = u + (size_t)M_DIM * H_DIM;    // 4*1280
    float* yglu = y63 + (size_t)B_DIM * H_DIM;  // 4*1280

    init_u_kernel<<<(M_DIM * H_DIM + 255) / 256, 256, 0, stream>>>(bb, u);

    dim3 g(2, 10, NKC);
    gemm_u_kernel<<<g, 256, 0, stream>>>(x, Wb, u);

    s4d_conv_kernel<<<(H_DIM + 255) / 256, 256, 0, stream>>>(u, log_dt, C, log_A_real, Dv, y63);
    zglu_kernel<<<(H_DIM + 255) / 256, 256, 0, stream>>>(Wc, bc, y63, yglu);
    head_kernel<<<1, 256, 0, stream>>>(yglu, W1, b1, W2, b2, out);
}
